// CausalSelfAttention_51599737094906
// MI455X (gfx1250) — compile-verified, matrix operand order checked by reading, not run on hardware
//
#include <hip/hip_runtime.h>
#include <hip/hip_bf16.h>
#include <stdint.h>

// CausalSelfAttention for MI455X (gfx1250, wave32, WMMA bf16).
// B=256, T=256, C=384, NH=8, HD=48.
// Workspace: 5*SQ + TC*C + C*C bf16 elements  (~253 MB).

#define B_  256
#define T_  256
#define C_  384
#define NH_ 8
#define HD_ 48
#define TC_ (3 * C_)   // 1152

typedef __attribute__((ext_vector_type(16))) __bf16 v16bf;
typedef __attribute__((ext_vector_type(8)))  float  v8f;

// One WMMA 16-bit operand fragment: 16 bf16 elements per lane (8 VGPRs).
// Element e holds K = 16*(e/8) + 8*(lane>=16) + (e%8) for row (A) /
// column (B) = lane%16, per the CDNA5 16-bit A/B layouts. q[0]/q[1] are the
// two contiguous 8-element K segments.
union Frag {
    v16bf v;
    uint4 q[2];
    unsigned short s[16];
};

__device__ __forceinline__ unsigned short f2bf(float f) {
    unsigned int u = __builtin_bit_cast(unsigned int, f);
    unsigned int r = u + 0x7FFFu + ((u >> 16) & 1u);   // round-to-nearest-even
    return (unsigned short)(r >> 16);
}

__device__ __forceinline__ v8f wmma_bf16(const Frag& a, const Frag& b, v8f c) {
    return __builtin_amdgcn_wmma_f32_16x16x32_bf16(false, a.v, false, b.v,
                                                   (short)0, c, false, false);
}

// ---------------------------------------------------------------------------
// Pre-pass: one-time f32 -> bf16 conversion (8 elements / thread).
// Keeps all rounding VALU out of the GEMM hot loops.
// ---------------------------------------------------------------------------
__global__ __launch_bounds__(256) void cvt_bf16(
    const float* __restrict__ in, unsigned short* __restrict__ out, int n8)
{
    const int i = blockIdx.x * blockDim.x + threadIdx.x;
    if (i >= n8) return;
    const float4* p = (const float4*)(in + (size_t)i * 8);
    const float4 f0 = p[0], f1 = p[1];
    const float fl[8] = {f0.x, f0.y, f0.z, f0.w, f1.x, f1.y, f1.z, f1.w};
    union { uint4 q; unsigned short s[8]; } u;
#pragma unroll
    for (int e = 0; e < 8; ++e) u.s[e] = f2bf(fl[e]);
    *(uint4*)(out + (size_t)i * 8) = u.q;
}

// ---------------------------------------------------------------------------
// Kernel 1: qkv = x @ attn_w^T + attn_b, all-bf16 operands, double-buffered.
// Wave strip: 16 rows x 64 cols (4 acc tiles); block = 8 waves stacked in M.
// Epilogue scatters Q,K head-major and V transposed (bf16).
// ---------------------------------------------------------------------------
__global__ __launch_bounds__(256) void qkv_gemm(
    const unsigned short* __restrict__ xbf, // [B*T, C] bf16
    const unsigned short* __restrict__ wbf, // [3C, C] bf16
    const float* __restrict__ bias,         // [3C]
    unsigned short* __restrict__ qws,       // [B*NH, T, HD]
    unsigned short* __restrict__ kws,       // [B*NH, T, HD]
    unsigned short* __restrict__ vtws)      // [B*NH, HD, T]
{
    const int lane = threadIdx.x & 31;
    const int wave = threadIdx.x >> 5;
    const int lo = lane & 15, hi = lane >> 4;

    const int row0 = blockIdx.x * 128 + wave * 16;
    const int col0 = blockIdx.y * 64;

    const unsigned short* xrow = xbf + (size_t)(row0 + lo) * C_;
    const unsigned short* wrow[4];
#pragma unroll
    for (int ct = 0; ct < 4; ++ct)
        wrow[ct] = wbf + (size_t)(col0 + ct * 16 + lo) * C_;

    v8f acc[4];
#pragma unroll
    for (int ct = 0; ct < 4; ++ct) {
        const float bv = bias[col0 + ct * 16 + lo];
#pragma unroll
        for (int i = 0; i < 8; ++i) acc[ct][i] = bv;
    }

    Frag a[2], b[2][4];
    {   // prologue load of k-step 0
        const int k0 = 8 * hi, k1 = 16 + 8 * hi;
        a[0].q[0] = *(const uint4*)(xrow + k0);
        a[0].q[1] = *(const uint4*)(xrow + k1);
#pragma unroll
        for (int ct = 0; ct < 4; ++ct) {
            b[0][ct].q[0] = *(const uint4*)(wrow[ct] + k0);
            b[0][ct].q[1] = *(const uint4*)(wrow[ct] + k1);
        }
    }
#pragma unroll
    for (int i = 0; i < 12; ++i) {          // C_/32 == 12
        const int cur = i & 1, nxt = cur ^ 1;
        if (i + 1 < 12) {                   // prefetch next k-step
            const int ks = (i + 1) * 32;
            const int k0 = ks + 8 * hi, k1 = ks + 16 + 8 * hi;
            a[nxt].q[0] = *(const uint4*)(xrow + k0);
            a[nxt].q[1] = *(const uint4*)(xrow + k1);
#pragma unroll
            for (int ct = 0; ct < 4; ++ct) {
                b[nxt][ct].q[0] = *(const uint4*)(wrow[ct] + k0);
                b[nxt][ct].q[1] = *(const uint4*)(wrow[ct] + k1);
            }
        }
#pragma unroll
        for (int ct = 0; ct < 4; ++ct)
            acc[ct] = wmma_bf16(a[cur], b[cur][ct], acc[ct]);
    }

    // Epilogue: C/D layout -> M = r + 8*hi, N = lane%16.
#pragma unroll
    for (int ct = 0; ct < 4; ++ct) {
        const int col = col0 + ct * 16 + lo;
        const int sec = col / C_;      // 0=Q, 1=K, 2=V
        const int cc  = col % C_;
        const int h   = cc / HD_;
        const int d   = cc % HD_;
#pragma unroll
        for (int r = 0; r < 8; ++r) {
            const int row = row0 + r + 8 * hi;   // token index in [0, B*T)
            const int bb = row / T_, t = row % T_;
            const size_t bh = (size_t)(bb * NH_ + h);
            const unsigned short val = f2bf(acc[ct][r]);
            if (sec == 0)      qws[(bh * T_ + t) * HD_ + d] = val;
            else if (sec == 1) kws[(bh * T_ + t) * HD_ + d] = val;
            else               vtws[(bh * HD_ + d) * T_ + t] = val;
        }
    }
}

// ---------------------------------------------------------------------------
// Kernel 2: causal attention for one (batch, head). Each wave handles 16
// queries; block = 8 waves = 128 queries; 2 blocks per head. Causal skip of
// key tiles above the diagonal halves the WMMA work.
// ---------------------------------------------------------------------------
__global__ __launch_bounds__(256) void attn_kernel(
    const unsigned short* __restrict__ qws,
    const unsigned short* __restrict__ kws,
    const unsigned short* __restrict__ vtws,
    unsigned short* __restrict__ attws)   // [B*T, C] bf16
{
    __shared__ __align__(16) unsigned short probs[8][16 * 256];  // 64 KB

    const int lane = threadIdx.x & 31;
    const int wave = threadIdx.x >> 5;
    const int lo = lane & 15, hi = lane >> 4;

    const int bh    = blockIdx.x >> 1;
    const int qbase = (blockIdx.x & 1) * 128 + wave * 16;

    const unsigned short* Qh  = qws  + (size_t)bh * T_ * HD_;
    const unsigned short* Kh  = kws  + (size_t)bh * T_ * HD_;
    const unsigned short* Vth = vtws + (size_t)bh * HD_ * T_;

    // Q fragments: 16 queries x 64 K (HD=48 zero-padded to 64).
    Frag aq[2];
#pragma unroll
    for (int kc = 0; kc < 2; ++kc) {
        const unsigned short* qrow = Qh + (size_t)(qbase + lo) * HD_;
        const int k0 = kc * 32 + 8 * hi;
        const int k1 = kc * 32 + 16 + 8 * hi;
        aq[kc].q[0] = (k0 + 8 <= HD_) ? *(const uint4*)(qrow + k0)
                                      : make_uint4(0, 0, 0, 0);
        aq[kc].q[1] = (k1 + 8 <= HD_) ? *(const uint4*)(qrow + k1)
                                      : make_uint4(0, 0, 0, 0);
    }

    // Scores: only key tiles at/below the diagonal (qbase is 16-aligned).
    const int nkt = (qbase >> 4) + 1;
    v8f st[16];
#pragma unroll
    for (int t = 0; t < 16; ++t) {
#pragma unroll
        for (int i = 0; i < 8; ++i) st[t][i] = 0.0f;
        if (t < nkt) {
#pragma unroll
            for (int kc = 0; kc < 2; ++kc) {
                const unsigned short* krow = Kh + (size_t)(t * 16 + lo) * HD_;
                const int k0 = kc * 32 + 8 * hi;
                const int k1 = kc * 32 + 16 + 8 * hi;
                Frag b;
                b.q[0] = (k0 + 8 <= HD_) ? *(const uint4*)(krow + k0)
                                         : make_uint4(0, 0, 0, 0);
                b.q[1] = (k1 + 8 <= HD_) ? *(const uint4*)(krow + k1)
                                         : make_uint4(0, 0, 0, 0);
                st[t] = wmma_bf16(aq[kc], b, st[t]);
            }
        }
    }

    // Row softmax. Row M = r + 8*hi lives in acc element r across one
    // 16-lane half; xor-shuffles with masks <16 stay inside the half.
    const float scale = 0.14433756729740643f;   // 1/sqrt(48)
    float inv[8];
#pragma unroll
    for (int r = 0; r < 8; ++r) {
        const int qg = qbase + r + 8 * hi;
        float mx = -3.0e38f;
#pragma unroll
        for (int t = 0; t < 16; ++t) {
            if (t < nkt) {
                const int kg = t * 16 + lo;
                float v = st[t][r] * scale;
                if (kg > qg) v = -3.0e38f;      // causal mask
                st[t][r] = v;
                mx = fmaxf(mx, v);
            }
        }
#pragma unroll
        for (int off = 8; off >= 1; off >>= 1)
            mx = fmaxf(mx, __shfl_xor(mx, off, 32));
        float sum = 0.0f;
#pragma unroll
        for (int t = 0; t < 16; ++t) {
            if (t < nkt) {
                const float e = __expf(st[t][r] - mx);
                st[t][r] = e;
                sum += e;
            }
        }
#pragma unroll
        for (int off = 8; off >= 1; off >>= 1)
            sum += __shfl_xor(sum, off, 32);
        inv[r] = 1.0f / sum;                    // fold 1/sum into the output
        const int m = r + 8 * hi;
#pragma unroll
        for (int t = 0; t < 16; ++t)
            probs[wave][m * 256 + t * 16 + lo] =
                (t < nkt) ? f2bf(st[t][r]) : (unsigned short)0;
    }

    // O = E @ V (unnormalized): E back from LDS in A-layout (ds_load_b128),
    // V^T rows as B operand; 3 d-tiles of 16, K = keys in chunks of 32.
    v8f o[3];
#pragma unroll
    for (int dt = 0; dt < 3; ++dt)
#pragma unroll
        for (int i = 0; i < 8; ++i) o[dt][i] = 0.0f;

    const int nchunk = (nkt + 1) >> 1;
#pragma unroll
    for (int c = 0; c < 8; ++c) {
        if (c < nchunk) {
            const int k0 = c * 32 + 8 * hi;
            const int k1 = c * 32 + 16 + 8 * hi;
            Frag a;
            const unsigned short* prow = &probs[wave][lo * 256];
            a.q[0] = *(const uint4*)(prow + k0);
            a.q[1] = *(const uint4*)(prow + k1);
#pragma unroll
            for (int dt = 0; dt < 3; ++dt) {
                const unsigned short* vrow = Vth + (size_t)(dt * 16 + lo) * T_;
                Frag b;
                b.q[0] = *(const uint4*)(vrow + k0);
                b.q[1] = *(const uint4*)(vrow + k1);
                o[dt] = wmma_bf16(a, b, o[dt]);
            }
        }
    }

    // Write normalized output, merged back to [B*T, C] bf16 for the proj GEMM.
    const int bb = bh >> 3;   // NH_ == 8
    const int h  = bh & 7;
#pragma unroll
    for (int dt = 0; dt < 3; ++dt) {
#pragma unroll
        for (int r = 0; r < 8; ++r) {
            const int m = r + 8 * hi;
            const int t = qbase + m;
            attws[((size_t)(bb * T_ + t)) * C_ + h * HD_ + dt * 16 + lo] =
                f2bf(o[dt][r] * inv[r]);
        }
    }
}

// ---------------------------------------------------------------------------
// Kernel 3: out = att @ proj_w^T + proj_b (f32 output), double-buffered.
// ---------------------------------------------------------------------------
__global__ __launch_bounds__(256) void proj_gemm(
    const unsigned short* __restrict__ att, // [B*T, C] bf16
    const unsigned short* __restrict__ wbf, // [C, C] bf16
    const float* __restrict__ bias,         // [C]
    float* __restrict__ out)                // [B*T, C] f32
{
    const int lane = threadIdx.x & 31;
    const int wave = threadIdx.x >> 5;
    const int lo = lane & 15, hi = lane >> 4;

    const int row0 = blockIdx.x * 128 + wave * 16;
    const int col0 = blockIdx.y * 64;

    const unsigned short* arow = att + (size_t)(row0 + lo) * C_;
    const unsigned short* wrow[4];
#pragma unroll
    for (int ct = 0; ct < 4; ++ct)
        wrow[ct] = wbf + (size_t)(col0 + ct * 16 + lo) * C_;

    v8f acc[4];
#pragma unroll
    for (int ct = 0; ct < 4; ++ct) {
        const float bv = bias[col0 + ct * 16 + lo];
#pragma unroll
        for (int i = 0; i < 8; ++i) acc[ct][i] = bv;
    }

    Frag a[2], b[2][4];
    {
        const int k0 = 8 * hi, k1 = 16 + 8 * hi;
        a[0].q[0] = *(const uint4*)(arow + k0);
        a[0].q[1] = *(const uint4*)(arow + k1);
#pragma unroll
        for (int ct = 0; ct < 4; ++ct) {
            b[0][ct].q[0] = *(const uint4*)(wrow[ct] + k0);
            b[0][ct].q[1] = *(const uint4*)(wrow[ct] + k1);
        }
    }
#pragma unroll
    for (int i = 0; i < 12; ++i) {
        const int cur = i & 1, nxt = cur ^ 1;
        if (i + 1 < 12) {
            const int ks = (i + 1) * 32;
            const int k0 = ks + 8 * hi, k1 = ks + 16 + 8 * hi;
            a[nxt].q[0] = *(const uint4*)(arow + k0);
            a[nxt].q[1] = *(const uint4*)(arow + k1);
#pragma unroll
            for (int ct = 0; ct < 4; ++ct) {
                b[nxt][ct].q[0] = *(const uint4*)(wrow[ct] + k0);
                b[nxt][ct].q[1] = *(const uint4*)(wrow[ct] + k1);
            }
        }
#pragma unroll
        for (int ct = 0; ct < 4; ++ct)
            acc[ct] = wmma_bf16(a[cur], b[cur][ct], acc[ct]);
    }

#pragma unroll
    for (int ct = 0; ct < 4; ++ct) {
        const int col = col0 + ct * 16 + lo;
#pragma unroll
        for (int r = 0; r < 8; ++r) {
            const int row = row0 + r + 8 * hi;
            out[(size_t)row * C_ + col] = acc[ct][r];
        }
    }
}

// ---------------------------------------------------------------------------
extern "C" void kernel_launch(void* const* d_in, const int* in_sizes, int n_in,
                              void* d_out, int out_size, void* d_ws, size_t ws_size,
                              hipStream_t stream) {
    (void)in_sizes; (void)n_in; (void)out_size; (void)ws_size;
    const float* x      = (const float*)d_in[0];
    const float* attn_w = (const float*)d_in[1];
    const float* attn_b = (const float*)d_in[2];
    const float* proj_w = (const float*)d_in[3];
    const float* proj_b = (const float*)d_in[4];
    float* out = (float*)d_out;

    const size_t SQ = (size_t)B_ * NH_ * T_ * HD_;   // 25,165,824 (== B*T*C)
    unsigned short* qws   = (unsigned short*)d_ws;
    unsigned short* kws   = qws + SQ;
    unsigned short* vtws  = kws + SQ;
    unsigned short* attws = vtws + SQ;
    unsigned short* xbf   = attws + SQ;              // [B*T, C]
    unsigned short* wbf   = xbf + SQ;                // [3C, C]
    unsigned short* pwbf  = wbf + (size_t)TC_ * C_;  // [C, C]

    dim3 blk(256);

    // One-shot f32 -> bf16 conversions (keeps rounding VALU off the hot path).
    const int nx8 = (int)(SQ / 8);
    const int nw8 = (TC_ * C_) / 8;
    const int np8 = (C_ * C_) / 8;
    cvt_bf16<<<dim3((nx8 + 255) / 256), blk, 0, stream>>>(x, xbf, nx8);
    cvt_bf16<<<dim3((nw8 + 255) / 256), blk, 0, stream>>>(attn_w, wbf, nw8);
    cvt_bf16<<<dim3((np8 + 255) / 256), blk, 0, stream>>>(proj_w, pwbf, np8);

    qkv_gemm<<<dim3((B_ * T_) / 128, TC_ / 64), blk, 0, stream>>>(
        xbf, wbf, attn_b, qws, kws, vtws);
    attn_kernel<<<dim3(B_ * NH_ * 2), blk, 0, stream>>>(qws, kws, vtws, attws);
    proj_gemm<<<dim3((B_ * T_) / 128, C_ / 64), blk, 0, stream>>>(
        attws, proj_w ? pwbf : pwbf, proj_b, out);
}
